// minLSTMCell_25907242730141
// MI455X (gfx1250) — matrix operand order repaired and checked by reading, not compiled, not run on hardware
//
#include <hip/hip_runtime.h>
#include <hip/hip_bf16.h>
#include <math.h>

typedef __bf16 bf16_t;
typedef __bf16 v8bf  __attribute__((ext_vector_type(8)));
typedef __bf16 v16bf __attribute__((ext_vector_type(16)));
typedef float  v8f   __attribute__((ext_vector_type(8)));

#define B_ 4
#define S_ 2048
#define D_ 1024
#define E_ 1536
#define K_ 4
#define M_ (B_*S_)   // 8192 rows for all GEMMs

enum { EPI_RES = 0, EPI_LOGSIG = 1, EPI_LOGG = 2, EPI_GELU = 3, EPI_BIAS = 4 };

// ---------------------------------------------------------------- utilities
__global__ void cast_f32_bf16(const float* __restrict__ in, bf16_t* __restrict__ out, int n) {
  int i = blockIdx.x * blockDim.x + threadIdx.x;
  if (i < n) out[i] = (bf16_t)in[i];
}

// Causal depthwise conv over time (K=4) + bias, output bf16 (B*S, D)
__global__ __launch_bounds__(256)
void dwconv_kernel(const float* __restrict__ x, const float* __restrict__ w,
                   const float* __restrict__ bias, bf16_t* __restrict__ out) {
  const int d   = blockIdx.y * blockDim.x + threadIdx.x;
  const int row = blockIdx.x;                 // b*S + s
  const int s   = row & (S_ - 1);
  float acc = bias[d];
#pragma unroll
  for (int k = 0; k < K_; ++k) {
    int sk = s - (K_ - 1) + k;
    if (sk >= 0) acc += x[(size_t)(row - (K_ - 1) + k) * D_ + d] * w[d * K_ + k];
  }
  out[(size_t)row * D_ + d] = (bf16_t)acc;
}

__device__ __forceinline__ float neg_softplus_neg(float z) {
  // -softplus(-z), numerically stable
  return -log1pf(expf(-fabsf(z))) - fmaxf(-z, 0.0f);
}

// ---------------------------------------------------------------- WMMA GEMM
// C(M x N) = A(M x Kd, bf16 row-major) * W(N x Kd, bf16 row-major)^T  + epilogue
// Double-buffered LDS; A tile staged with async global->LDS (ASYNCcnt),
// B tile staged through registers with an LDS transpose scatter.
template <int EPI>
__global__ __launch_bounds__(256)
void gemm_bf16_wmma(const bf16_t* __restrict__ A, const bf16_t* __restrict__ W,
                    const float* __restrict__ bias, const float* __restrict__ res,
                    void* __restrict__ outp, int N, int Kd) {
  __shared__ __align__(16) bf16_t At[2][128][40];    // 128 M x 32 K (padded)
  __shared__ __align__(16) bf16_t Bt[2][32][136];    // 32 K x 128 N (padded)

  const int t     = threadIdx.x;
  const int lane  = t & 31;
  const int wave  = t >> 5;
  const int waveM = wave >> 2;                       // 0..1  (64 rows each)
  const int waveN = wave & 3;                        // 0..3  (32 cols each)
  const int mBase = blockIdx.y * 128;
  const int nBase = blockIdx.x * 128;

  const int lr = t >> 1;                             // 0..127 : tile row to load
  const int lc = (t & 1) * 16;                       // 0 / 16 : K-chunk

  const bf16_t* gA = A + (size_t)(mBase + lr) * Kd + lc;
  const bf16_t* gW = W + (size_t)(nBase + lr) * Kd + lc;

  v8f acc[4][2];
#pragma unroll
  for (int i = 0; i < 4; ++i)
#pragma unroll
    for (int j = 0; j < 2; ++j)
#pragma unroll
      for (int r = 0; r < 8; ++r) acc[i][j][r] = 0.0f;

  // --- async stage of one A-tile 32B chunk (two b128 async loads) ---
  auto asyncA = [&](int buf, int k0) {
    unsigned l0 = (unsigned)(size_t)&At[buf][lr][lc];
    const bf16_t* g0 = gA + k0;
    asm volatile("global_load_async_to_lds_b128 %0, %1, off\n\t"
                 "global_load_async_to_lds_b128 %2, %3, off"
                 :: "v"(l0), "v"(g0), "v"(l0 + 16u), "v"(g0 + 8)
                 : "memory");
  };
  auto loadBregs = [&](int k0, v8bf& w0, v8bf& w1) {
    const bf16_t* src = gW + k0;
    w0 = *(const v8bf*)src;
    w1 = *(const v8bf*)(src + 8);
  };
  auto scatterB = [&](int buf, v8bf w0, v8bf w1) {
#pragma unroll
    for (int j = 0; j < 8; ++j) Bt[buf][lc + j][lr]     = w0[j];
#pragma unroll
    for (int j = 0; j < 8; ++j) Bt[buf][lc + 8 + j][lr] = w1[j];
  };

  // prologue: stage tile 0
  {
    asyncA(0, 0);
    v8bf w0, w1;
    loadBregs(0, w0, w1);
    scatterB(0, w0, w1);
    asm volatile("s_wait_asynccnt 0x0" ::: "memory");
    __syncthreads();
  }

  int buf = 0;
  for (int k0 = 0; k0 < Kd; k0 += 32) {
    const int kn = k0 + 32;
    const bool pre = (kn < Kd);
    v8bf w0n, w1n;
    if (pre) {                       // prefetch next slice while computing
      asyncA(buf ^ 1, kn);
      loadBregs(kn, w0n, w1n);
    }

    // B fragments: lane&15 = N column, lane>>4 picks K-half (same K order as A)
    v16bf bfr[2];
#pragma unroll
    for (int fn = 0; fn < 2; ++fn) {
      const int n0 = waveN * 32 + fn * 16 + (lane & 15);
#pragma unroll
      for (int s = 0; s < 16; ++s)
        bfr[fn][s] = Bt[buf][(s < 8 ? 0 : 16) + (lane >> 4) * 8 + (s & 7)][n0];
    }

    // A fragments: lane&15 = M row, lane>>4 picks K-half
    const int am = lane & 15;
    const int c0 = (lane >> 4) * 8;
#pragma unroll
    for (int fm = 0; fm < 4; ++fm) {
      const int mrow = waveM * 64 + fm * 16 + am;
      union { v16bf v; v8bf h[2]; } u;
      u.h[0] = *(const v8bf*)&At[buf][mrow][c0];
      u.h[1] = *(const v8bf*)&At[buf][mrow][c0 + 16];
      acc[fm][0] = __builtin_amdgcn_wmma_f32_16x16x32_bf16(
          false, u.v, false, bfr[0], (short)0, acc[fm][0], false, false);
      acc[fm][1] = __builtin_amdgcn_wmma_f32_16x16x32_bf16(
          false, u.v, false, bfr[1], (short)0, acc[fm][1], false, false);
    }

    if (pre) {
      scatterB(buf ^ 1, w0n, w1n);
      asm volatile("s_wait_asynccnt 0x0" ::: "memory");
    }
    __syncthreads();                 // single barrier per K-slice
    buf ^= 1;
  }

  // --- epilogue: C layout = VGPR r, lanes 0-15 -> M=r,N=lane; 16-31 -> M=r+8 ---
  const int nLane = lane & 15;
  const int mOff  = (lane < 16) ? 0 : 8;
#pragma unroll
  for (int fm = 0; fm < 4; ++fm) {
#pragma unroll
    for (int fn = 0; fn < 2; ++fn) {
      const int ncol = nBase + waveN * 32 + fn * 16 + nLane;
#pragma unroll
      for (int r = 0; r < 8; ++r) {
        const int mrow = mBase + waveM * 64 + fm * 16 + mOff + r;
        float v = acc[fm][fn][r];
        if (bias) v += bias[ncol];
        if (res)  v += res[(size_t)mrow * N + ncol];
        if (EPI == EPI_LOGSIG) {
          v = neg_softplus_neg(v);
        } else if (EPI == EPI_LOGG) {
          v = (v >= 0.0f) ? logf(v + 0.5f) : neg_softplus_neg(v);
        } else if (EPI == EPI_GELU) {
          v = 0.5f * v * (1.0f + erff(v * 0.70710678118654752f));
        }
        const size_t oi = (size_t)mrow * N + ncol;
        if (EPI == EPI_GELU) ((bf16_t*)outp)[oi] = (bf16_t)v;
        else                 ((float*)outp)[oi]  = v;
      }
    }
  }
}

// ---------------------------------------------------------------- LayerNorm
__global__ __launch_bounds__(256)
void layernorm_kernel(const float* __restrict__ x, const float* __restrict__ g,
                      const float* __restrict__ b, bf16_t* __restrict__ out) {
  __shared__ float red[256];
  const int row = blockIdx.x;
  const float* xr = x + (size_t)row * D_;
  float vals[4];
  float s = 0.0f;
#pragma unroll
  for (int i = 0; i < 4; ++i) { vals[i] = xr[threadIdx.x + i * 256]; s += vals[i]; }
  red[threadIdx.x] = s; __syncthreads();
  for (int off = 128; off > 0; off >>= 1) {
    if ((int)threadIdx.x < off) red[threadIdx.x] += red[threadIdx.x + off];
    __syncthreads();
  }
  const float mu = red[0] * (1.0f / D_);
  __syncthreads();
  float vs = 0.0f;
#pragma unroll
  for (int i = 0; i < 4; ++i) { float d0 = vals[i] - mu; vs += d0 * d0; }
  red[threadIdx.x] = vs; __syncthreads();
  for (int off = 128; off > 0; off >>= 1) {
    if ((int)threadIdx.x < off) red[threadIdx.x] += red[threadIdx.x + off];
    __syncthreads();
  }
  const float inv = rsqrtf(red[0] * (1.0f / D_) + 1e-5f);
#pragma unroll
  for (int i = 0; i < 4; ++i) {
    const int d0 = threadIdx.x + i * 256;
    out[(size_t)row * D_ + d0] = (bf16_t)((vals[i] - mu) * inv * g[d0] + b[d0]);
  }
}

// ---------------------------------------------------------------- minLSTM scan
// log_h_t = logaddexp(log_f_t + log_h_{t-1}, log_i_t + log_th_t); h = exp(log_h)
__global__ __launch_bounds__(256)
void scan_kernel(const float* __restrict__ lf, const float* __restrict__ li,
                 const float* __restrict__ lth, bf16_t* __restrict__ h) {
  const int gid = blockIdx.x * blockDim.x + threadIdx.x;  // b*E + e
  const int b = gid / E_;
  const int e = gid - b * E_;
  size_t idx = (size_t)b * S_ * E_ + e;
  float lh = -0.69314718055994531f;  // log(0.5)
  for (int t = 0; t < S_; ++t, idx += E_) {
    const float a = lf[idx] + lh;
    const float c = li[idx] + lth[idx];
    const float m = fmaxf(a, c);
    lh = m + log1pf(expf(-fabsf(a - c)));
    h[idx] = (bf16_t)expf(lh);
  }
}

// ---------------------------------------------------------------- launch
extern "C" void kernel_launch(void* const* d_in, const int* in_sizes, int n_in,
                              void* d_out, int out_size, void* d_ws, size_t ws_size,
                              hipStream_t stream) {
  (void)in_sizes; (void)n_in; (void)out_size; (void)ws_size;
  const float* x     = (const float*)d_in[0];
  const float* dw_w  = (const float*)d_in[1];
  const float* dw_b  = (const float*)d_in[2];
  const float* pw_b  = (const float*)d_in[4];
  const float* ln1g  = (const float*)d_in[5];
  const float* ln1b  = (const float*)d_in[6];
  const float* ln2g  = (const float*)d_in[11];
  const float* ln2b  = (const float*)d_in[12];
  const float* b1    = (const float*)d_in[14];
  const float* b2    = (const float*)d_in[16];

  char* ws = (char*)d_ws;
  size_t cur = 0;
  auto alloc = [&](size_t bytes) { size_t r = cur; cur += (bytes + 255) & ~(size_t)255; return r; };

  const size_t M = M_;
  bf16_t* dwb  = (bf16_t*)(ws + alloc(M * D_ * 2));       // conv output (bf16)
  float*  x12  = (float*)(ws + alloc(M * D_ * 4));        // x1, later x2
  bf16_t* xn   = (bf16_t*)(ws + alloc(M * D_ * 2));       // xn1, later xn2
  size_t  lgfO = alloc(M * E_ * 4);
  float*  lgf  = (float*)(ws + lgfO);
  float*  lgi  = (float*)(ws + alloc(M * E_ * 4));
  float*  lgth = (float*)(ws + alloc(M * E_ * 4));
  bf16_t* hbuf = (bf16_t*)(ws + alloc(M * E_ * 2));
  bf16_t* mid  = (bf16_t*)(ws + lgfO);                    // MLP mid (64MB) reuses dead lgf+lgi (96MB)
  bf16_t* pwwb = (bf16_t*)(ws + alloc((size_t)D_ * D_ * 2));
  bf16_t* wfb  = (bf16_t*)(ws + alloc((size_t)E_ * D_ * 2));
  bf16_t* wib  = (bf16_t*)(ws + alloc((size_t)E_ * D_ * 2));
  bf16_t* whb  = (bf16_t*)(ws + alloc((size_t)E_ * D_ * 2));
  bf16_t* wdb  = (bf16_t*)(ws + alloc((size_t)D_ * E_ * 2));
  bf16_t* w1b  = (bf16_t*)(ws + alloc((size_t)4 * D_ * D_ * 2));
  bf16_t* w2b  = (bf16_t*)(ws + alloc((size_t)D_ * 4 * D_ * 2));

  auto cast = [&](const void* src, bf16_t* dst, int n) {
    cast_f32_bf16<<<(n + 255) / 256, 256, 0, stream>>>((const float*)src, dst, n);
  };
  cast(d_in[3],  pwwb, D_ * D_);
  cast(d_in[7],  wfb,  E_ * D_);
  cast(d_in[8],  wib,  E_ * D_);
  cast(d_in[9],  whb,  E_ * D_);
  cast(d_in[10], wdb,  D_ * E_);
  cast(d_in[13], w1b,  4 * D_ * D_);
  cast(d_in[15], w2b,  D_ * 4 * D_);

  // 1) depthwise causal conv
  dwconv_kernel<<<dim3((unsigned)M, D_ / 256), 256, 0, stream>>>(x, dw_w, dw_b, dwb);

  // 2) pointwise conv GEMM + bias + residual(x) -> x1 (fp32)
  dim3 gD(D_ / 128, (unsigned)M / 128);
  gemm_bf16_wmma<EPI_RES><<<gD, 256, 0, stream>>>(dwb, pwwb, pw_b, x, x12, D_, D_);

  // 3) LayerNorm1 -> xn1 (bf16)
  layernorm_kernel<<<(unsigned)M, 256, 0, stream>>>(x12, ln1g, ln1b, xn);

  // 4) gate projections with fused log-space activations
  dim3 gE(E_ / 128, (unsigned)M / 128);
  gemm_bf16_wmma<EPI_LOGSIG><<<gE, 256, 0, stream>>>(xn, wfb, nullptr, nullptr, lgf,  E_, D_);
  gemm_bf16_wmma<EPI_LOGSIG><<<gE, 256, 0, stream>>>(xn, wib, nullptr, nullptr, lgi,  E_, D_);
  gemm_bf16_wmma<EPI_LOGG>  <<<gE, 256, 0, stream>>>(xn, whb, nullptr, nullptr, lgth, E_, D_);

  // 5) log-space sequential scan -> h (bf16)
  scan_kernel<<<(B_ * E_) / 256, 256, 0, stream>>>(lgf, lgi, lgth, hbuf);

  // 6) down-projection GEMM + residual(x) -> x2 (fp32, reuses x1 buffer)
  gemm_bf16_wmma<EPI_RES><<<gD, 256, 0, stream>>>(hbuf, wdb, nullptr, x, x12, D_, E_);

  // 7) LayerNorm2 -> xn2 (bf16, reuses xn buffer)
  layernorm_kernel<<<(unsigned)M, 256, 0, stream>>>(x12, ln2g, ln2b, xn);

  // 8) MLP up + exact GELU -> mid (bf16)
  dim3 gU(4 * D_ / 128, (unsigned)M / 128);
  gemm_bf16_wmma<EPI_GELU><<<gU, 256, 0, stream>>>(xn, w1b, b1, nullptr, mid, 4 * D_, D_);

  // 9) MLP down + bias -> d_out (fp32)
  gemm_bf16_wmma<EPI_BIAS><<<gD, 256, 0, stream>>>(mid, w2b, b2, nullptr, d_out, D_, 4 * D_);
}
